// LNN_82884278878211
// MI455X (gfx1250) — matrix operand order repaired and checked by reading, not compiled, not run on hardware
//
#include <hip/hip_runtime.h>

typedef __attribute__((ext_vector_type(16))) _Float16 v16h;
typedef __attribute__((ext_vector_type(8)))  _Float16 v8h;
typedef __attribute__((ext_vector_type(4)))  _Float16 v4h;
typedef __attribute__((ext_vector_type(2)))  _Float16 v2h;
typedef __attribute__((ext_vector_type(8)))  float    v8f;

#define NP 22                      // 1 value + 6 grad + 15 Hessian planes
#define SW_HALFS   (25*1024)       // 25 weight slots, each 32x32 f16 [n][k]
#define SW_BYTES   (SW_HALFS*2)    // 51200
#define SBIAS_N    (18*32)
#define SBIAS_BYTES (SBIAS_N*4)    // 2304
#define WAVE_BYTES 102784          // U,Z,T(32w) + P2,Q,Y(16w) + Vbuf
#define LDS_BYTES  (SW_BYTES + SBIAS_BYTES + 2*WAVE_BYTES)  // 259072 <= 320KB

// weight slot metadata: K, N, src-is-[n][k] (fc weights), relu
__device__ __constant__ int dK[25] = {3,3,3,3,32,32,32,3,32,32,32,32,32,3,32,32,32,32,3,32,32,3,32,32,32};
__device__ __constant__ int dN[25] = {32,32,3,32,32,32,32,32,3,32,32,32,32,32,3,32,1,32,1,3,1,32,32,32,1};
__device__ __constant__ int dT[25] = {0,0,0,0,0,0,0,0,0,0,0,0,0,0,0,0,0,0,0,0,0,1,1,1,1};
__device__ __constant__ int dR[25] = {0,0,0,0,0,1,0,0,0,0,0,1,0,0,0,0,1,0,0,0,0,0,0,0,0};
__device__ __constant__ int dBsz[18] = {3,32,32,32,3,32,32,32,32,3,32,32,3,1,32,32,32,1};
// Hessian plane p (0..14) <-> pair (i,j), i in {3,4,5}, j <= i
__device__ __constant__ int cPI[15] = {3,3,3,3,4,4,4,4,4,5,5,5,5,5,5};
__device__ __constant__ int cPJ[15] = {0,1,2,3,0,1,2,3,4,0,1,2,3,4,5};

struct PtrPack { const float* w[25]; const float* b[18]; };

// ---------------- prep: pad/transpose/relu weights + biases into ws ----------
__global__ void lnn_prep(PtrPack P, _Float16* __restrict__ wsH, float* __restrict__ wsB) {
  int blk = blockIdx.x, t = threadIdx.x;
  if (blk < 25) {
    int n = t >> 5, k = t & 31;
    float v = 0.f;
    if (k < dK[blk] && n < dN[blk]) {
      v = dT[blk] ? P.w[blk][n*dK[blk] + k] : P.w[blk][k*dN[blk] + n];
      if (dR[blk]) v = fmaxf(v, 0.f);
    }
    wsH[blk*1024 + n*32 + k] = (_Float16)v;
  } else if (t < SBIAS_N) {
    int s = t >> 5, i = t & 31;
    wsB[t] = (i < dBsz[s]) ? P.b[s][i] : 0.f;
  }
}

// ---------------- device helpers --------------------------------------------
__device__ __forceinline__ float sp_f(float x)  { return (x > 20.f) ? x : log1pf(__expf(x)); }
__device__ __forceinline__ float sig_f(float x) { return 1.f/(1.f + __expf(-x)); }

// A-operand, plane-major [p][m][k] (plane stride 16*aF halfs).
// Lane holds row m; K chunks {c0..c0+7, c0+16..c0+23} -> two ds_load_b128.
__device__ __forceinline__ v16h ldA(const _Float16* a, int aF, int p, int m, int c0) {
  const _Float16* base = a + (p*16 + m)*aF;
  v8h lo = *(const v8h*)(base + c0);
  v8h hi = {0,0,0,0,0,0,0,0};
  if (aF > 16) hi = *(const v8h*)(base + c0 + 16);
  return __builtin_shufflevector(lo, hi, 0,1,2,3,4,5,6,7,8,9,10,11,12,13,14,15);
}

struct Term { const _Float16* a; int aF; int ws; };

// 22-plane GEMM pass: dst[p][m][n] = sum_t A_t @ W_t  (+bias on value plane).
// Both n-tiles share a single A-operand load; two independent WMMA chains.
template<int NT, int NTILES>
__device__ void mm_pass(_Float16* dst, int dStride,
                        const Term* terms, const _Float16* sW, const float* sB,
                        int bslot, float* vout, int lane)
{
  int lm = lane & 15, half = lane >> 4;
  v16h B0[NT], B1[NT];
  #pragma unroll
  for (int t = 0; t < NT; ++t) {
    { const _Float16* b0p = sW + terms[t].ws*1024 + lm*32 + half*8;
      B0[t] = __builtin_shufflevector(*(const v8h*)b0p, *(const v8h*)(b0p + 16), 0,1,2,3,4,5,6,7,8,9,10,11,12,13,14,15); }
    if (NTILES > 1) {
      const _Float16* b1p = sW + terms[t].ws*1024 + (lm + 16)*32 + half*8;
      B1[t] = __builtin_shufflevector(*(const v8h*)b1p, *(const v8h*)(b1p + 16), 0,1,2,3,4,5,6,7,8,9,10,11,12,13,14,15);
    }
  }
  float bias0 = sB[bslot*32 + lm];
  float bias1 = (NTILES > 1) ? sB[bslot*32 + lm + 16] : 0.f;

  #pragma unroll 2
  for (int p = 0; p < NP; ++p) {
    v8f c0 = {}, c1 = {};
    #pragma unroll
    for (int t = 0; t < NT; ++t) {
      v16h a = ldA(terms[t].a, terms[t].aF, p, lm, half*8);
      c0 = __builtin_amdgcn_wmma_f32_16x16x32_f16(false, a, false, B0[t],
                                                  (short)0, c0, false, false);
      if (NTILES > 1)
        c1 = __builtin_amdgcn_wmma_f32_16x16x32_f16(false, a, false, B1[t],
                                                    (short)0, c1, false, false);
    }
    if (p == 0) {
      #pragma unroll
      for (int r = 0; r < 8; ++r) { c0[r] += bias0; if (NTILES > 1) c1[r] += bias1; }
    }
    if (vout) {                       // V-path epilogue: f32 column 0 only
      if (lm == 0) {
        #pragma unroll
        for (int r = 0; r < 8; ++r) vout[(r + half*8)*NP + p] = c0[r];
      }
    } else {                          // C layout: m = r + 8*half, n = lm (+16)
      #pragma unroll
      for (int r = 0; r < 8; ++r) {
        dst[(p*16 + r + half*8)*dStride + lm] = (_Float16)c0[r];
        if (NTILES > 1)
          dst[(p*16 + r + half*8)*dStride + lm + 16] = (_Float16)c1[r];
      }
    }
  }
}

// softplus with 2nd-order chain rule, in place; 4 positions per iteration
__device__ void sp_pass(_Float16* buf, int F, int lane) {
  int PS = 16*F;               // plane stride in halfs
  int nch = PS/4;
  for (int ch = lane; ch < nch; ch += 32) {
    _Float16* p0 = buf + ch*4;
    v4h in[NP], outv[NP];
    #pragma unroll
    for (int i = 0; i < NP; ++i) in[i] = *(const v4h*)(p0 + i*PS);
    #pragma unroll
    for (int j = 0; j < 4; ++j) {
      float v[NP];
      #pragma unroll
      for (int i = 0; i < NP; ++i) v[i] = (float)in[i][j];
      float d1 = sig_f(v[0]);
      float d2 = d1*(1.f - d1);
      outv[0][j] = (_Float16)sp_f(v[0]);
      #pragma unroll
      for (int q = 0; q < 6; ++q) outv[1+q][j] = (_Float16)(d1*v[1+q]);
      #pragma unroll
      for (int p = 0; p < 15; ++p)
        outv[7+p][j] = (_Float16)(d1*v[7+p] + d2*v[1+cPI[p]]*v[1+cPJ[p]]);
    }
    #pragma unroll
    for (int i = 0; i < NP; ++i) *(v4h*)(p0 + i*PS) = outv[i];
  }
}

// tuple Hadamard product: dst <- src (.) dst; 2 positions per iteration
__device__ void prod_pass(_Float16* dst, const _Float16* src, int F, int lane) {
  int PS = 16*F;
  int nch = PS/2;
  for (int ch = lane; ch < nch; ch += 32) {
    _Float16* d0 = dst + ch*2;
    const _Float16* s0 = src + ch*2;
    v2h av[NP], bv[NP], ov[NP];
    #pragma unroll
    for (int i = 0; i < NP; ++i) { av[i] = *(const v2h*)(s0 + i*PS); bv[i] = *(const v2h*)(d0 + i*PS); }
    #pragma unroll
    for (int j = 0; j < 2; ++j) {
      float a[NP], b[NP];
      #pragma unroll
      for (int i = 0; i < NP; ++i) { a[i] = (float)av[i][j]; b[i] = (float)bv[i][j]; }
      ov[0][j] = (_Float16)(a[0]*b[0]);
      #pragma unroll
      for (int q = 0; q < 6; ++q) ov[1+q][j] = (_Float16)(a[1+q]*b[0] + a[0]*b[1+q]);
      #pragma unroll
      for (int p = 0; p < 15; ++p)
        ov[7+p][j] = (_Float16)(a[7+p]*b[0] + a[0]*b[7+p]
                              + a[1+cPI[p]]*b[1+cPJ[p]] + a[1+cPJ[p]]*b[1+cPI[p]]);
    }
    #pragma unroll
    for (int i = 0; i < NP; ++i) *(v2h*)(d0 + i*PS) = ov[i];
  }
}

// seed tuples, plane-major 16-wide: one b128 store per plane per lane.
// lane -> (m = lane/2, n0 = 8*(lane&1)); only n<3 columns are non-zero.
__device__ void seed_pass(_Float16* Q, _Float16* Y, const float* x, int base, int lane) {
  int m = lane >> 1, n0 = (lane & 1)*8;
  float xr[6];
  #pragma unroll
  for (int i = 0; i < 6; ++i) xr[i] = x[(base+m)*6 + i];
  v8h z = {0,0,0,0,0,0,0,0};
  v8h qv = z, yv = z;
  if (n0 == 0) {
    #pragma unroll
    for (int j = 0; j < 3; ++j) { qv[j] = (_Float16)xr[j]; yv[j] = (_Float16)xr[3+j]; }
  }
  *(v8h*)(Q + m*16 + n0) = qv;
  *(v8h*)(Y + m*16 + n0) = yv;
  #pragma unroll
  for (int j = 0; j < 6; ++j) {
    v8h qg = z, yg = z;
    if (n0 == 0) {
      if (j < 3)  qg[j]   = (_Float16)1.f;
      if (j >= 3) yg[j-3] = (_Float16)1.f;
    }
    *(v8h*)(Q + (1+j)*256 + m*16 + n0) = qg;
    *(v8h*)(Y + (1+j)*256 + m*16 + n0) = yg;
  }
  #pragma unroll
  for (int p = 0; p < 15; ++p) {
    *(v8h*)(Q + (7+p)*256 + m*16 + n0) = z;
    *(v8h*)(Y + (7+p)*256 + m*16 + n0) = z;
  }
}

// ---------------- main kernel -----------------------------------------------
__global__ void __launch_bounds__(64)
lnn_main(const float* __restrict__ x, const _Float16* __restrict__ wsH,
         const float* __restrict__ wsB, float* __restrict__ out, int nTiles)
{
  extern __shared__ char smemRaw[];
  _Float16* sW = (_Float16*)smemRaw;
  float*    sB = (float*)(smemRaw + SW_BYTES);
  int wave = threadIdx.x >> 5, lane = threadIdx.x & 31;

  for (int i = threadIdx.x; i < SW_HALFS; i += blockDim.x) sW[i] = wsH[i];
  for (int i = threadIdx.x; i < SBIAS_N;  i += blockDim.x) sB[i] = wsB[i];

  char* wb = smemRaw + SW_BYTES + SBIAS_BYTES + wave*WAVE_BYTES;
  _Float16* U  = (_Float16*)(wb);            // 22 x 16x32 f16, plane-major
  _Float16* Z  = (_Float16*)(wb + 22528);
  _Float16* T  = (_Float16*)(wb + 45056);
  _Float16* P2 = (_Float16*)(wb + 67584);    // 22 x 16x16 f16, plane-major
  _Float16* Q  = (_Float16*)(wb + 78848);
  _Float16* Y  = (_Float16*)(wb + 90112);
  float*    Vb = (float*)   (wb + 101376);   // 16x22 f32

  int tile = blockIdx.x*2 + wave;
  int base = tile*16;
  if (tile < nTiles)
    __builtin_prefetch(x + base*6, 0, 0);    // global_prefetch_b8 for the tile's inputs
  __syncthreads();
  if (tile >= nTiles) return;

  seed_pass(Q, Y, x, base, lane);

  Term t1[3];
  // ---- V(q) path (uses Q, scratch U; result into Vb) ----
  t1[0] = {Q,16,21}; mm_pass<1,2>(U,32,t1,sW,sB,14,nullptr,lane); sp_pass(U,32,lane);
  t1[0] = {U,32,22}; mm_pass<1,2>(U,32,t1,sW,sB,15,nullptr,lane); sp_pass(U,32,lane);
  t1[0] = {U,32,23}; mm_pass<1,2>(U,32,t1,sW,sB,16,nullptr,lane); sp_pass(U,32,lane);
  t1[0] = {U,32,24}; mm_pass<1,1>(nullptr,32,t1,sW,sB,17,Vb,lane);

  // ---- layer 0: u1 = sp(q@w_h0+b_h0); z1 = sp((y*(q@w_yu0+b_y0))@w_y0 + q@w_u0 + b_0) ----
  t1[0] = {Q,16,2};  mm_pass<1,1>(P2,16,t1,sW,sB,0,nullptr,lane);
  prod_pass(P2, Y, 16, lane);
  t1[0] = {Q,16,0};  mm_pass<1,2>(U,32,t1,sW,sB,1,nullptr,lane); sp_pass(U,32,lane);
  t1[0] = {P2,16,1}; t1[1] = {Q,16,3};
  mm_pass<2,2>(Z,32,t1,sW,sB,2,nullptr,lane); sp_pass(Z,32,lane);

  // ---- layer 1 ----
  t1[0] = {U,32,6};  mm_pass<1,2>(T,32,t1,sW,sB,5,nullptr,lane);
  prod_pass(T, Z, 32, lane);                       // z1 * (u1@w_zu1+b_z1)
  t1[0] = {U,32,8};  mm_pass<1,1>(P2,16,t1,sW,sB,4,nullptr,lane);
  prod_pass(P2, Y, 16, lane);                      // y * (u1@w_yu1+b_y1)
  t1[0] = {T,32,5}; t1[1] = {P2,16,7}; t1[2] = {U,32,9};
  mm_pass<3,2>(Z,32,t1,sW,sB,6,nullptr,lane); sp_pass(Z,32,lane);   // z2
  t1[0] = {U,32,4};  mm_pass<1,2>(U,32,t1,sW,sB,3,nullptr,lane); sp_pass(U,32,lane); // u2

  // ---- layer 2 ----
  t1[0] = {U,32,12}; mm_pass<1,2>(T,32,t1,sW,sB,8,nullptr,lane);
  prod_pass(T, Z, 32, lane);
  t1[0] = {U,32,14}; mm_pass<1,1>(P2,16,t1,sW,sB,9,nullptr,lane);
  prod_pass(P2, Y, 16, lane);
  t1[0] = {T,32,11}; t1[1] = {P2,16,13}; t1[2] = {U,32,15};
  mm_pass<3,2>(Z,32,t1,sW,sB,10,nullptr,lane); sp_pass(Z,32,lane);  // z3
  t1[0] = {U,32,10}; mm_pass<1,2>(U,32,t1,sW,sB,7,nullptr,lane); sp_pass(U,32,lane); // u3

  // ---- output head: zout preact into Z col 0 (stride 32, single n-tile) ----
  t1[0] = {U,32,17}; mm_pass<1,2>(T,32,t1,sW,sB,11,nullptr,lane);
  prod_pass(T, Z, 32, lane);
  t1[0] = {U,32,19}; mm_pass<1,1>(P2,16,t1,sW,sB,12,nullptr,lane);
  prod_pass(P2, Y, 16, lane);
  t1[0] = {T,32,16}; t1[1] = {P2,16,18}; t1[2] = {U,32,20};
  mm_pass<3,1>(Z,32,t1,sW,sB,13,nullptr,lane);

  // ---- per-sample finish: softplus(zout) - V, extract g/C/M, 3x3 solve ----
  if (lane < 16) {
    int m = lane;
    float v[NP], Lt[NP];
    #pragma unroll
    for (int p = 0; p < NP; ++p) v[p] = (float)Z[p*512 + m*32];   // column 0, plane-major
    float d1 = sig_f(v[0]), d2 = d1*(1.f - d1);
    Lt[0] = sp_f(v[0]) - Vb[m*NP + 0];
    #pragma unroll
    for (int j = 0; j < 6; ++j) Lt[1+j] = d1*v[1+j] - Vb[m*NP + 1 + j];
    #pragma unroll
    for (int p = 0; p < 15; ++p)
      Lt[7+p] = d1*v[7+p] + d2*v[1+cPI[p]]*v[1+cPJ[p]] - Vb[m*NP + 7 + p];

    const int Ci[3][3] = {{0,1,2},{4,5,6},{9,10,11}};   // H[3+i][j], j<3
    float dq0 = x[(base+m)*6 + 3], dq1 = x[(base+m)*6 + 4], dq2 = x[(base+m)*6 + 5];
    float r0 = Lt[1] - (dq0*Lt[7+Ci[0][0]] + dq1*Lt[7+Ci[0][1]] + dq2*Lt[7+Ci[0][2]]);
    float r1 = Lt[2] - (dq0*Lt[7+Ci[1][0]] + dq1*Lt[7+Ci[1][1]] + dq2*Lt[7+Ci[1][2]]);
    float r2 = Lt[3] - (dq0*Lt[7+Ci[2][0]] + dq1*Lt[7+Ci[2][1]] + dq2*Lt[7+Ci[2][2]]);
    float M00 = Lt[7+3],  M01 = Lt[7+7],  M02 = Lt[7+12];
    float M11 = Lt[7+8],  M12 = Lt[7+13], M22 = Lt[7+14];
    float a00 = M11*M22 - M12*M12, a01 = M02*M12 - M01*M22, a02 = M01*M12 - M02*M11;
    float a11 = M00*M22 - M02*M02, a12 = M01*M02 - M00*M12, a22 = M00*M11 - M01*M01;
    float id  = 1.f/(M00*a00 + M01*a01 + M02*a02);
    float dd0 = (a00*r0 + a01*r1 + a02*r2)*id;
    float dd1 = (a01*r0 + a11*r1 + a12*r2)*id;
    float dd2 = (a02*r0 + a12*r1 + a22*r2)*id;
    int g = base + m;
    out[g*6+0] = x[g*6+0] + 0.001f*dq0;
    out[g*6+1] = x[g*6+1] + 0.001f*dq1;
    out[g*6+2] = x[g*6+2] + 0.001f*dq2;
    out[g*6+3] = dq0 + 0.001f*dd0;
    out[g*6+4] = dq1 + 0.001f*dd1;
    out[g*6+5] = dq2 + 0.001f*dd2;
  }
}

// ---------------- host launcher ---------------------------------------------
extern "C" void kernel_launch(void* const* d_in, const int* in_sizes, int n_in,
                              void* d_out, int out_size, void* d_ws, size_t ws_size,
                              hipStream_t stream)
{
  PtrPack P;
  for (int i = 0; i < 21; ++i) P.w[i] = (const float*)d_in[1+i];
  P.w[21] = (const float*)d_in[36];  // fc1_w
  P.w[22] = (const float*)d_in[38];  // fc2_w
  P.w[23] = (const float*)d_in[40];  // fc3_w
  P.w[24] = (const float*)d_in[42];  // fc4_w
  for (int i = 0; i < 14; ++i) P.b[i] = (const float*)d_in[22+i];
  P.b[14] = (const float*)d_in[37];  // fc1_b
  P.b[15] = (const float*)d_in[39];
  P.b[16] = (const float*)d_in[41];
  P.b[17] = (const float*)d_in[43];

  _Float16* wsH = (_Float16*)d_ws;
  float*    wsB = (float*)((char*)d_ws + SW_BYTES);
  lnn_prep<<<26, 1024, 0, stream>>>(P, wsH, wsB);

  (void)hipFuncSetAttribute((const void*)lnn_main,
                            hipFuncAttributeMaxDynamicSharedMemorySize, LDS_BYTES);

  const float* xp = (const float*)d_in[0];
  int nSamples = in_sizes[0] / 6;
  int nTiles   = (nSamples + 15) / 16;
  int nBlocks  = (nTiles + 1) / 2;        // 2 waves (2 tiles) per workgroup
  lnn_main<<<nBlocks, 64, LDS_BYTES, stream>>>(xp, wsH, wsB, (float*)d_out, nTiles);
}
